// ImageBEVGaussianEncoder_49417893708292
// MI455X (gfx1250) — matrix operand order repaired and checked by reading, not compiled, not run on hardware
//
#include <hip/hip_runtime.h>
#include <hip/hip_bf16.h>
#include <stdint.h>

typedef __attribute__((ext_vector_type(16))) _Float16 v16h;
typedef __attribute__((ext_vector_type(8)))  float    v8f;
typedef unsigned int u32;
typedef __attribute__((ext_vector_type(4))) u32 u32x4;
typedef __attribute__((ext_vector_type(8))) int  i32x8;
typedef __attribute__((ext_vector_type(4))) int  i32x4;

#define BATCH 4
#define NYC 256
#define NXC 256
#define SCELL (NYC*NXC)
#define HF 64
#define WF 96
#define NPIX (BATCH*HF*WF)

// ---------------- workspace layout (bytes) ----------------
#define OFF_IMG16   ((size_t)0)                 // 37,748,736  (4*3*1024*1536 f16)
#define OFF_FEAT    ((size_t)0)                 // reuse after conv1: 24576*64 f32
#define OFF_SW9     ((size_t)6291456)           // 24576*9 f32
#define OFF_IDX9    ((size_t)7176192)           // 24576*9 i32
#define OFF_X1      ((size_t)37748736)          // 4*32*512*768 f16
#define OFF_CANVAS  ((size_t)37748736)          // reuse after conv2: 4*65536*64 f32
#define OFF_WACC    ((size_t)104857600)         // 4*65536 f32
#define OFF_X2      ((size_t)138412032)         // 4*64*256*384 f16
#define OFF_X3      ((size_t)188743680)         // 4*128*128*192 f16
#define OFF_X4      ((size_t)213909504)         // 4*128*64*96 f16
#define OFF_FH      ((size_t)220200960)         // 4*128*64*96 f16
#define OFF_WP1     ((size_t)226492416)         // 32*32 f16
#define OFF_WP2     (OFF_WP1 + 2048)            // 288*64 f16
#define OFF_WP3     (OFF_WP2 + 36864)           // 576*128 f16
#define OFF_WP4     (OFF_WP3 + 147456)          // 1152*128 f16
#define OFF_WPF1    (OFF_WP4 + 294912)          // 1152*128 f16
#define OFF_WPF2    (OFF_WPF1 + 294912)         // 128*64 f16

// ---------------- utility kernels ----------------
__global__ void f32_to_f16_kernel(const float* __restrict__ in, _Float16* __restrict__ out, int n) {
    int i = blockIdx.x * blockDim.x + threadIdx.x;
    if (i < n) out[i] = (_Float16)in[i];
}

__global__ void zero_f32_kernel(float* __restrict__ p, int n) {
    int i = blockIdx.x * blockDim.x + threadIdx.x;
    if (i < n) p[i] = 0.0f;
}

// OIHW f32 -> [Kpad][Cout] f16, zero padded for k >= Ktot
__global__ void pack_w_kernel(const float* __restrict__ w, _Float16* __restrict__ wp,
                              int Cin, int Cout, int khw, int Ktot, int Kpad) {
    int i = blockIdx.x * blockDim.x + threadIdx.x;
    if (i >= Kpad * Cout) return;
    int k = i / Cout, co = i % Cout;
    float v = 0.0f;
    if (k < Ktot) {
        int c = k / khw, r = k % khw;
        v = w[(co * Cin + c) * khw + r];
    }
    wp[k * Cout + co] = (_Float16)v;
}

// ---------------- TDM: load 32 x NTW f16 weight tile into LDS ----------------
// D# per CDNA5 ISA. 2D tensor: row length Cout (f16),
// row stride Cout, tile = NTW x 32 rows, linear packed into LDS at lds_off.
__device__ __forceinline__ void tdm_load_b_tile(const _Float16* gsrc, u32 lds_off,
                                                int Cout, int NTW) {
    uint64_t ga = (uint64_t)(uintptr_t)gsrc;
    u32x4 g0;
    g0[0] = 1u;                                              // count=1 (user descriptor)
    g0[1] = lds_off;                                         // lds_addr (bytes)
    g0[2] = (u32)ga;                                         // global_addr[31:0]
    g0[3] = (u32)((ga >> 32) & 0x01ffffffu) | 0x80000000u;   // global_addr[56:32] | type=2
    i32x8 g1;
    g1[0] = 0x00010000;                                      // workgroup_mask=0, data_size=1 (2B)
    g1[1] = (int)(((u32)Cout & 0xffffu) << 16);              // tensor_dim0[15:0]
    g1[2] = (int)((((u32)Cout >> 16) & 0xffffu) | (32u << 16)); // tensor_dim0[31:16] | tensor_dim1[15:0]=32
    g1[3] = (int)(((u32)NTW & 0xffffu) << 16);               // tensor_dim1[31:16]=0 | tile_dim0=NTW
    g1[4] = 32;                                              // tile_dim1=32, tile_dim2=0
    g1[5] = Cout;                                            // tensor_dim0_stride[31:0]
    g1[6] = 0;                                               // stride0[47:32] | stride1[15:0]
    g1[7] = 0;                                               // stride1[47:16]
    i32x4 gz4 = {0, 0, 0, 0};
    i32x8 gz8 = {0, 0, 0, 0, 0, 0, 0, 0};
    __builtin_amdgcn_tensor_load_to_lds(g0, g1, gz4, gz4, gz8, 0);
}

// ---------------- implicit-GEMM conv via WMMA ----------------
// NT waves per block. Block computes 16(pixels) x NT*16(out-channels).
// A tile (16x32 f16) staged cooperatively (gather); B tile (32 x NT*16 f16)
// DMA'd by the Tensor Data Mover. mode 0: relu(acc*scale+bias) -> f16 NCHW.
// mode 1: acc+bias -> f32 NHWC (features).
template<int NT>
__global__ void __launch_bounds__(NT * 32) conv_wmma_kernel(
    const _Float16* __restrict__ in, const _Float16* __restrict__ wp,
    const float* __restrict__ scale, const float* __restrict__ bias,
    _Float16* __restrict__ out16, float* __restrict__ outf32,
    int Cin, int Hin, int Win, int Cout, int Hout, int Wout,
    int khw, int ksw, int pad, int stride, int Ktot, int Kpad, int mode)
{
    constexpr int NTW = NT * 16;            // B tile width
    constexpr int CS  = 16 / NT;            // A elems staged per thread
    __shared__ _Float16 smem[512 + NT * 512];   // As @ byte 0, Bs @ byte 1024
    _Float16* As = smem;                    // [16][32] row-major
    _Float16* Bs = smem + 512;              // [32][NTW] row-major

    const int t    = threadIdx.x;
    const int wv   = t >> 5;                // wave id: owns N-slice wv*16
    const int lane = t & 31;
    const int grp  = (lane >> 4) & 1;
    const int lm   = lane & 15;

    const int HW = Hout * Wout;
    const int Mtot = BATCH * HW;
    const int tm  = blockIdx.x * 16;        // pixel tile base
    const int tn0 = blockIdx.y * NTW;       // out-channel tile base

    // per-thread A row: one output pixel, decomposed ONCE
    const int mRow = t & 15;
    const int kk0  = (t >> 4) * CS;
    const int pixA = tm + mRow;
    const bool pvalid = pixA < Mtot;
    int ab = 0, aoy = 0, aox = 0;
    if (pvalid) {
        ab = pixA / HW;
        int rem = pixA % HW;
        aoy = rem / Wout;
        aox = rem % Wout;
    }
    const int iy0 = aoy * stride - pad;
    const int ix0 = aox * stride - pad;
    const size_t inB = (size_t)ab * Cin;

    v8f acc = {};

    for (int kb = 0; kb < Kpad; kb += 32) {
        // ---- issue TDM weight-tile DMA (wave 0 only) ----
        if (t < 32) {
            tdm_load_b_tile(wp + (size_t)kb * Cout + tn0, 1024u, Cout, NTW);
        }
        // ---- stage A tile (gather from NCHW f16) ----
        #pragma unroll
        for (int i = 0; i < CS; ++i) {
            int kk = kk0 + i;
            int k = kb + kk;
            _Float16 hv = (_Float16)0.0f;
            if (pvalid && k < Ktot) {
                int c = k / khw, r = k % khw;
                int dy = r / ksw, dx = r % ksw;
                int iy = iy0 + dy, ix = ix0 + dx;
                if (iy >= 0 && iy < Hin && ix >= 0 && ix < Win)
                    hv = in[((inB + c) * Hin + iy) * Win + ix];
            }
            As[mRow * 32 + kk] = hv;
        }
        if (t < 32) __builtin_amdgcn_s_wait_tensorcnt(0);
        __syncthreads();

        // ---- build fragments per CDNA5 wave32 layouts ----
        v16h afr, bfr;
        #pragma unroll
        for (int j = 0; j < 8; ++j) {
            int kb2 = ((j & 3) * 2) + grp * 8 + ((j >> 2) * 16);
            afr[2 * j]     = As[lm * 32 + kb2];
            afr[2 * j + 1] = As[lm * 32 + kb2 + 1];
        }
        const int ncol = wv * 16 + lm;
        #pragma unroll
        for (int j = 0; j < 8; ++j) {
            int kb2 = ((j & 3) * 2) + grp * 8 + ((j >> 2) * 16);
            bfr[2 * j]     = Bs[kb2 * NTW + ncol];
            bfr[2 * j + 1] = Bs[(kb2 + 1) * NTW + ncol];
        }

        acc = __builtin_amdgcn_wmma_f32_16x16x32_f16(
            false, afr, false, bfr, (short)0, acc, false, false);

        __syncthreads();
    }

    // ---- epilogue: C/D layout (VGPR j -> M = j + 8*grp, N = lm) ----
    const int co = tn0 + wv * 16 + lm;
    #pragma unroll
    for (int j = 0; j < 8; ++j) {
        int m = j + grp * 8;
        int pix = tm + m;
        if (pix < Mtot) {
            if (mode == 0) {
                float v = acc[j] * scale[co] + bias[co];
                v = fmaxf(v, 0.0f);
                int b = pix / HW, rem = pix % HW;
                int oy = rem / Wout, ox = rem % Wout;
                out16[((size_t)(b * Cout + co) * Hout + oy) * Wout + ox] = (_Float16)v;
            } else {
                outf32[(size_t)pix * Cout + co] = acc[j] + bias[co];
            }
        }
    }
}

// ---------------- lift: depth/opacity heads + backprojection + tap weights ----------------
__global__ void lift_kernel(const _Float16* __restrict__ x4,
                            const float* __restrict__ cam_K, const float* __restrict__ T_lc,
                            const float* __restrict__ dw, const float* __restrict__ dbias,
                            const float* __restrict__ ow, const float* __restrict__ obias,
                            const int* __restrict__ img_h, const int* __restrict__ img_w,
                            float* __restrict__ sw9, int* __restrict__ idx9)
{
    int p = blockIdx.x * blockDim.x + threadIdx.x;
    if (p >= NPIX) return;
    const int HW = HF * WF;
    int b = p / HW, rem = p % HW, y = rem / WF, x = rem % WF;

    float dl[16];
    #pragma unroll
    for (int d = 0; d < 16; ++d) dl[d] = dbias[d];
    float ol = obias[0];
    for (int c = 0; c < 128; ++c) {
        float xv = (float)x4[((size_t)(b * 128 + c) * HF + y) * WF + x];
        ol += ow[c] * xv;
        #pragma unroll
        for (int d = 0; d < 16; ++d) dl[d] += dw[d * 128 + c] * xv;
    }
    float op = 1.0f / (1.0f + __expf(-ol));

    float mx = dl[0];
    #pragma unroll
    for (int d = 1; d < 16; ++d) mx = fmaxf(mx, dl[d]);
    float se = 0.0f, sz = 0.0f;
    #pragma unroll
    for (int d = 0; d < 16; ++d) {
        float e = __expf(dl[d] - mx);
        se += e;
        sz += e * (1.0f + (float)d * (59.0f / 15.0f));   // linspace(1,60,16)
    }
    float z = sz / se;

    float ih = (float)img_h[0], iw = (float)img_w[0];
    float ys = ((float)y + 0.5f) * (ih / (float)HF);
    float xs = ((float)x + 0.5f) * (iw / (float)WF);
    const float* K = cam_K + b * 9;
    float fx = fmaxf(K[0], 1e-6f), fy = fmaxf(K[4], 1e-6f);
    float cx = K[2], cy = K[5];
    float xc = (xs - cx) * z / fx;
    float yc = (ys - cy) * z / fy;
    const float* T = T_lc + b * 16;
    float xw = T[0] * xc + T[1] * yc + T[2]  * z + T[3];
    float yw = T[4] * xc + T[5] * yc + T[6]  * z + T[7];
    float zw = T[8] * xc + T[9] * yc + T[10] * z + T[11];

    int xi = (int)floorf((xw + 51.2f) / 0.4f);
    int yi = (int)floorf((yw + 51.2f) / 0.4f);
    bool inb = (xi >= 0) && (xi < NXC) && (yi >= 0) && (yi < NYC) && (zw >= -5.0f) && (zw < 3.0f);
    float bw = ((op >= 0.05f) && inb) ? op : 0.0f;

    #pragma unroll
    for (int ti = 0; ti < 9; ++ti) {
        int dy = ti / 3 - 1, dx = ti % 3 - 1;
        int ty = yi + dy, tx = xi + dx;
        bool vm = (tx >= 0) && (tx < NXC) && (ty >= 0) && (ty < NYC);
        float kw = __expf(-(float)(dx * dx + dy * dy) / (2.0f * 0.8f * 0.8f));
        float sw = vm ? bw * kw : 0.0f;
        sw9[p * 9 + ti] = sw;
        idx9[p * 9 + ti] = (vm && sw > 0.0f) ? (b * SCELL + ty * NXC + tx) : -1;
    }
}

// ---------------- splat: 64 threads = 64 channels per pixel ----------------
__global__ void __launch_bounds__(64) splat_kernel(const float* __restrict__ feats,
                                                   const float* __restrict__ sw9,
                                                   const int* __restrict__ idx9,
                                                   float* __restrict__ canvas,
                                                   float* __restrict__ wacc)
{
    int p = blockIdx.x;
    int c = threadIdx.x;
    float fv = feats[(size_t)p * 64 + c];
    #pragma unroll
    for (int t = 0; t < 9; ++t) {
        int idx = idx9[p * 9 + t];
        if (idx >= 0) {
            float sw = sw9[p * 9 + t];
            atomicAdd(&canvas[(size_t)idx * 64 + c], fv * sw);
            if (c == 0) atomicAdd(&wacc[idx], sw);
        }
    }
}

// ---------------- normalize + NHWC->NCHW to d_out ----------------
__global__ void norm_kernel(const float* __restrict__ canvas, const float* __restrict__ wacc,
                            float* __restrict__ out)
{
    int i = blockIdx.x * blockDim.x + threadIdx.x;
    if (i >= BATCH * SCELL * 64) return;
    int c = i & 63;
    int cell = i >> 6;                 // b*SCELL + y*NXC + x
    float w = wacc[cell];
    float v = (w > 0.0f) ? canvas[(size_t)cell * 64 + c] / fmaxf(w, 1e-6f) : 0.0f;
    int b = cell >> 16;
    int yx = cell & 65535;
    out[(((size_t)(b * 64 + c)) << 16) + yx] = v;
}

// ---------------- host launch ----------------
extern "C" void kernel_launch(void* const* d_in, const int* in_sizes, int n_in,
                              void* d_out, int out_size, void* d_ws, size_t ws_size,
                              hipStream_t stream) {
    const float* images = (const float*)d_in[0];
    const float* cam_K  = (const float*)d_in[1];
    const float* T_lc   = (const float*)d_in[2];
    const float* w1 = (const float*)d_in[3];  const float* s1 = (const float*)d_in[4];  const float* b1 = (const float*)d_in[5];
    const float* w2 = (const float*)d_in[6];  const float* s2 = (const float*)d_in[7];  const float* b2 = (const float*)d_in[8];
    const float* w3 = (const float*)d_in[9];  const float* s3 = (const float*)d_in[10]; const float* b3 = (const float*)d_in[11];
    const float* w4 = (const float*)d_in[12]; const float* s4 = (const float*)d_in[13]; const float* b4 = (const float*)d_in[14];
    const float* fw1 = (const float*)d_in[15]; const float* fs1 = (const float*)d_in[16]; const float* fb1 = (const float*)d_in[17];
    const float* fw2 = (const float*)d_in[18]; const float* fbias2 = (const float*)d_in[19];
    const float* dw = (const float*)d_in[20];  const float* dbias = (const float*)d_in[21];
    const float* ow = (const float*)d_in[22];  const float* obias = (const float*)d_in[23];
    const int* img_h = (const int*)d_in[24];   const int* img_w = (const int*)d_in[25];

    char* ws = (char*)d_ws;
    _Float16* img16  = (_Float16*)(ws + OFF_IMG16);
    _Float16* x1     = (_Float16*)(ws + OFF_X1);
    _Float16* x2     = (_Float16*)(ws + OFF_X2);
    _Float16* x3     = (_Float16*)(ws + OFF_X3);
    _Float16* x4     = (_Float16*)(ws + OFF_X4);
    _Float16* fh     = (_Float16*)(ws + OFF_FH);
    float*    feats  = (float*)(ws + OFF_FEAT);
    float*    sw9    = (float*)(ws + OFF_SW9);
    int*      idx9   = (int*)(ws + OFF_IDX9);
    float*    canvas = (float*)(ws + OFF_CANVAS);
    float*    wacc   = (float*)(ws + OFF_WACC);
    _Float16* wp1  = (_Float16*)(ws + OFF_WP1);
    _Float16* wp2  = (_Float16*)(ws + OFF_WP2);
    _Float16* wp3  = (_Float16*)(ws + OFF_WP3);
    _Float16* wp4  = (_Float16*)(ws + OFF_WP4);
    _Float16* wpf1 = (_Float16*)(ws + OFF_WPF1);
    _Float16* wpf2 = (_Float16*)(ws + OFF_WPF2);
    float* out = (float*)d_out;

    // 1) convert images to f16
    {
        int n = BATCH * 3 * 1024 * 1536;
        f32_to_f16_kernel<<<(n + 255) / 256, 256, 0, stream>>>(images, img16, n);
    }
    // 2) pack weights
    auto pack = [&](const float* w, _Float16* wp, int Cin, int Cout, int khw, int Ktot, int Kpad) {
        int n = Kpad * Cout;
        pack_w_kernel<<<(n + 255) / 256, 256, 0, stream>>>(w, wp, Cin, Cout, khw, Ktot, Kpad);
    };
    pack(w1, wp1, 3, 32, 9, 27, 32);
    pack(w2, wp2, 32, 64, 9, 288, 288);
    pack(w3, wp3, 64, 128, 9, 576, 576);
    pack(w4, wp4, 128, 128, 9, 1152, 1152);
    pack(fw1, wpf1, 128, 128, 9, 1152, 1152);
    pack(fw2, wpf2, 128, 64, 1, 128, 128);

    // 3) conv stack
    auto conv = [&](const _Float16* in, const _Float16* wp, const float* sc, const float* bi,
                    _Float16* o16, float* of32,
                    int Cin, int Hin, int Win, int Cout, int Hout, int Wout,
                    int ks, int pad, int stride, int Ktot, int Kpad, int mode) {
        int Mtot = BATCH * Hout * Wout;
        if (Cout >= 64) {
            dim3 grid((Mtot + 15) / 16, Cout / 64);
            conv_wmma_kernel<4><<<grid, 128, 0, stream>>>(in, wp, sc, bi, o16, of32,
                Cin, Hin, Win, Cout, Hout, Wout, ks * ks, ks, pad, stride, Ktot, Kpad, mode);
        } else {
            dim3 grid((Mtot + 15) / 16, Cout / 32);
            conv_wmma_kernel<2><<<grid, 64, 0, stream>>>(in, wp, sc, bi, o16, of32,
                Cin, Hin, Win, Cout, Hout, Wout, ks * ks, ks, pad, stride, Ktot, Kpad, mode);
        }
    };
    conv(img16, wp1, s1, b1, x1, nullptr, 3, 1024, 1536, 32, 512, 768, 3, 1, 2, 27, 32, 0);
    conv(x1, wp2, s2, b2, x2, nullptr, 32, 512, 768, 64, 256, 384, 3, 1, 2, 288, 288, 0);
    conv(x2, wp3, s3, b3, x3, nullptr, 64, 256, 384, 128, 128, 192, 3, 1, 2, 576, 576, 0);
    conv(x3, wp4, s4, b4, x4, nullptr, 128, 128, 192, 128, 64, 96, 3, 1, 2, 1152, 1152, 0);
    conv(x4, wpf1, fs1, fb1, fh, nullptr, 128, 64, 96, 128, 64, 96, 3, 1, 1, 1152, 1152, 0);
    conv(fh, wpf2, nullptr, fbias2, nullptr, feats, 128, 64, 96, 64, 64, 96, 1, 0, 1, 128, 128, 1);

    // 4) lift
    lift_kernel<<<(NPIX + 255) / 256, 256, 0, stream>>>(x4, cam_K, T_lc, dw, dbias, ow, obias,
                                                        img_h, img_w, sw9, idx9);
    // 5) zero canvas + wacc (contiguous)
    {
        int n = BATCH * SCELL * 64 + BATCH * SCELL;
        zero_f32_kernel<<<(n + 255) / 256, 256, 0, stream>>>(canvas, n);
    }
    // 6) splat
    splat_kernel<<<NPIX, 64, 0, stream>>>(feats, sw9, idx9, canvas, wacc);
    // 7) normalize -> NCHW output
    {
        int n = BATCH * SCELL * 64;
        norm_kernel<<<(n + 255) / 256, 256, 0, stream>>>(canvas, wacc, out);
    }
}
